// CrossLayerLateral_69458211111444
// MI455X (gfx1250) — matrix operand order repaired and checked by reading, not compiled, not run on hardware
//
#include <hip/hip_runtime.h>

// ---------------------------------------------------------------------------
// out = x_current + alpha * (x_prev @ W^T), W densified (16MB) from COO.
// Compute-bound dense GEMM -> bf16 WMMA, f32 accumulate.
// Block tile: 256(M) x 64(N), K-step 32. 8 waves, each wave owns 32x64
// (8 accumulators) -> 8 back-to-back v_wmma per K-step.
// LDS uses a chunk-XOR swizzle (4-dword granularity) so both staging and
// fragment access are conflict-free b128 ops.
// Fast path: GLOBAL_LOAD_ASYNC_TO_LDS_B128 DMA into double-buffered LDS,
// one barrier per K-step, s_wait_asynccnt sync.
// Fallback: register double-buffer + ds_store_b128, two barriers.
// ws: W f32 16MB | W bf16 8MB | (optional) X bf16 32MB -> preconverted A.
// ---------------------------------------------------------------------------

#define H      2048
#define BM     256                // block tile M (8 waves x 32 rows)
#define TN     64                 // block tile N
#define TK     32                 // K step
#define ATILE  (BM * 16)          // A tile dwords (256 rows x 16 dwords)
#define BTILE  (TN * 16)          // B tile dwords

#if __has_builtin(__builtin_amdgcn_global_load_async_to_lds_b128)
#define HAS_ASYNC 1
#else
#define HAS_ASYNC 0
#endif

typedef __attribute__((ext_vector_type(16))) __bf16 v16bf;
typedef __attribute__((ext_vector_type(2)))  __bf16 v2bf;
typedef __attribute__((ext_vector_type(8)))  float  v8f;
typedef int v4i __attribute__((vector_size(16)));   // matches builtin param pointee

union FragBF {
    unsigned int u[8];
    uint4        u4[2];
    v16bf        v;
};

// chunk-XOR swizzled dword offset inside a tile (16 dwords per row, 4 chunks)
__device__ __forceinline__ int swz(int row, int chunk) {
    return row * 16 + ((chunk ^ ((row >> 2) & 3)) << 2);
}

__device__ __forceinline__ unsigned int pack2bf(float lo, float hi) {
#if __has_builtin(__builtin_amdgcn_cvt_pk_bf16_f32)
    v2bf p = __builtin_amdgcn_cvt_pk_bf16_f32(lo, hi);
    return __builtin_bit_cast(unsigned int, p);
#else
    unsigned int ul = __float_as_uint(lo);
    ul += 0x7FFFu + ((ul >> 16) & 1u);          // round-to-nearest-even
    unsigned int uh = __float_as_uint(hi);
    uh += 0x7FFFu + ((uh >> 16) & 1u);
    return (ul >> 16) | (uh & 0xFFFF0000u);
#endif
}

__device__ __forceinline__ void wait_async0() {
#if __has_builtin(__builtin_amdgcn_s_wait_asynccnt)
    __builtin_amdgcn_s_wait_asynccnt(0);
#else
    asm volatile("s_wait_asynccnt 0x0" ::: "memory");
#endif
}

// ---- 1) zero dense W (f32) -------------------------------------------------
__global__ void cl_zero_kernel(float4* __restrict__ Wf) {
    int t = blockIdx.x * blockDim.x + threadIdx.x;
    Wf[t] = make_float4(0.f, 0.f, 0.f, 0.f);
}

// ---- 2) COO scatter, duplicates summed ------------------------------------
__global__ void cl_scatter_kernel(const int* __restrict__ idx,
                                  const float* __restrict__ val,
                                  float* __restrict__ Wf, int nnz) {
    int t = blockIdx.x * blockDim.x + threadIdx.x;
    if (t < nnz) {
        int i = idx[t];          // row
        int j = idx[nnz + t];    // col
        atomicAdd(&Wf[i * H + j], val[t]);
    }
}

// ---- 3) convert f32 -> packed bf16 (used for W and optionally x_prev) ------
__global__ void cl_convert_kernel(const float4* __restrict__ src,
                                  uint2* __restrict__ dst) {
    int t = blockIdx.x * blockDim.x + threadIdx.x;
    float4 f = src[t];
    dst[t] = make_uint2(pack2bf(f.x, f.y), pack2bf(f.z, f.w));
}

// ---- 4) tiled bf16 WMMA GEMM + epilogue ------------------------------------
template <bool PRECONV>
__launch_bounds__(256)
__global__ void cl_gemm_kernel(const float* __restrict__ xprevF,        // f32 A (fallback)
                               const unsigned int* __restrict__ xprevB, // bf16 A dwords (fast)
                               const float* __restrict__ xcur,
                               const unsigned int* __restrict__ Wb,     // bf16 W[H][H] dwords
                               const float* __restrict__ alphap,
                               float* __restrict__ out) {
    constexpr bool ASYNCP = PRECONV && (HAS_ASYNC != 0);
    __shared__ __align__(16) unsigned int Ad[(ASYNCP ? 2 : 1) * ATILE];
    __shared__ __align__(16) unsigned int Bd[(ASYNCP ? 2 : 1) * BTILE];

    const int tid   = threadIdx.x;
    const int wave  = tid >> 5;
    const int lane  = tid & 31;
    const int half  = lane >> 4;     // 0: lanes 0-15, 1: lanes 16-31
    const int l15   = lane & 15;
    const int nBase = blockIdx.x * TN;
    const int mBase = blockIdx.y * BM;

    const float alpha = alphap[0];

    v8f acc[2][4] = {};              // [mi][s] : wave covers 32(M) x 64(N)

    // -- staging geometry: 512 (row,half) A tasks, 2 per thread; 256 B tasks --
    const int arow  = tid >> 1;                  // 0..127 ; second task row +128
    const int ahalf = tid & 1;                   // covers chunks 2h, 2h+1
    const int brow  = tid >> 2;                  // 0..63
    const int bq    = tid & 3;                   // one chunk

    const int aOff0  = swz(arow,        2 * ahalf);
    const int aOff0b = swz(arow,        2 * ahalf + 1);
    const int aOff1  = swz(arow + 128,  2 * ahalf);
    const int aOff1b = swz(arow + 128,  2 * ahalf + 1);
    const int bOff   = swz(brow, bq);

    const size_t aGI0 = (size_t)(mBase + arow) * (H / 2) + ahalf * 8;  // bf16-dword idx
    const size_t aGI1 = aGI0 + (size_t)128 * (H / 2);
    const size_t bGI  = (size_t)(nBase + brow) * (H / 2) + bq * 4;

    // -- loop-invariant fragment read offsets (swizzled, b128-aligned) --
    const int row0  = wave * 32 + l15;
    const int aF0lo = swz(row0,      half), aF0hi = swz(row0,      2 + half);
    const int aF1lo = swz(row0 + 16, half), aF1hi = swz(row0 + 16, 2 + half);
    int bFlo[4], bFhi[4];
#pragma unroll
    for (int s = 0; s < 4; ++s) {
        bFlo[s] = swz(s * 16 + l15, half);
        bFhi[s] = swz(s * 16 + l15, 2 + half);
    }

    auto compute = [&](const unsigned int* At, const unsigned int* Bt) {
        FragBF a[2], b[4];
        a[0].u4[0] = *(const uint4*)&At[aF0lo];
        a[0].u4[1] = *(const uint4*)&At[aF0hi];
        a[1].u4[0] = *(const uint4*)&At[aF1lo];
        a[1].u4[1] = *(const uint4*)&At[aF1hi];
#pragma unroll
        for (int s = 0; s < 4; ++s) {
            b[s].u4[0] = *(const uint4*)&Bt[bFlo[s]];
            b[s].u4[1] = *(const uint4*)&Bt[bFhi[s]];
        }
#pragma unroll
        for (int mi = 0; mi < 2; ++mi)
#pragma unroll
            for (int s = 0; s < 4; ++s)
                acc[mi][s] = __builtin_amdgcn_wmma_f32_16x16x32_bf16(
                    false, a[mi].v, false, b[s].v, (short)0, acc[mi][s], false, false);
    };

#if HAS_ASYNC
    if constexpr (ASYNCP) {
        // ---- async DMA path: global -> LDS, double-buffered, 1 barrier/iter ----
#define CL_ASYNC_LD(gp, lp)                                                  \
        __builtin_amdgcn_global_load_async_to_lds_b128(                      \
            (__attribute__((address_space(1))) v4i*)(gp),                    \
            (__attribute__((address_space(3))) v4i*)(lp), 0, 0)
        auto issue = [&](unsigned int* Ab, unsigned int* Bb, int k) {
            const size_t kd = (size_t)(k >> 1);
            CL_ASYNC_LD(xprevB + aGI0 + kd,     Ab + aOff0);
            CL_ASYNC_LD(xprevB + aGI0 + kd + 4, Ab + aOff0b);
            CL_ASYNC_LD(xprevB + aGI1 + kd,     Ab + aOff1);
            CL_ASYNC_LD(xprevB + aGI1 + kd + 4, Ab + aOff1b);
            CL_ASYNC_LD(Wb     + bGI  + kd,     Bb + bOff);
        };
        issue(Ad, Bd, 0);                       // prologue: tile 0 -> buffer 0
        int abuf = 0, bbuf = 0;
        for (int k0 = 0; k0 < H; k0 += TK) {
            wait_async0();                      // current tile landed (per wave)
            __syncthreads();                    // all waves' tiles landed
            const unsigned int* At = Ad + abuf;
            const unsigned int* Bt = Bd + bbuf;
            abuf ^= ATILE; bbuf ^= BTILE;
            if (k0 + TK < H) issue(Ad + abuf, Bd + bbuf, k0 + TK);
            compute(At, Bt);
        }
#undef CL_ASYNC_LD
    } else
#endif
    {
        // ---- sync fallback: register double-buffer + ds_store_b128 ----
        uint4  aRB[4], bR;
        float4 aRF[8];
        if constexpr (PRECONV) {
            aRB[0] = *(const uint4*)(xprevB + aGI0);
            aRB[1] = *(const uint4*)(xprevB + aGI0 + 4);
            aRB[2] = *(const uint4*)(xprevB + aGI1);
            aRB[3] = *(const uint4*)(xprevB + aGI1 + 4);
        } else {
            const float4* a0 = (const float4*)(xprevF + (size_t)(mBase + arow) * H + ahalf * 16);
            const float4* a1 = (const float4*)(xprevF + (size_t)(mBase + arow + 128) * H + ahalf * 16);
#pragma unroll
            for (int q = 0; q < 4; ++q) { aRF[q] = a0[q]; aRF[4 + q] = a1[q]; }
        }
        bR = *(const uint4*)(Wb + bGI);

        for (int k0 = 0; k0 < H; k0 += TK) {
            __syncthreads();
            if constexpr (PRECONV) {
                *(uint4*)&Ad[aOff0]  = aRB[0];
                *(uint4*)&Ad[aOff0b] = aRB[1];
                *(uint4*)&Ad[aOff1]  = aRB[2];
                *(uint4*)&Ad[aOff1b] = aRB[3];
            } else {
                *(uint4*)&Ad[aOff0]  = make_uint4(pack2bf(aRF[0].x, aRF[0].y), pack2bf(aRF[0].z, aRF[0].w),
                                                  pack2bf(aRF[1].x, aRF[1].y), pack2bf(aRF[1].z, aRF[1].w));
                *(uint4*)&Ad[aOff0b] = make_uint4(pack2bf(aRF[2].x, aRF[2].y), pack2bf(aRF[2].z, aRF[2].w),
                                                  pack2bf(aRF[3].x, aRF[3].y), pack2bf(aRF[3].z, aRF[3].w));
                *(uint4*)&Ad[aOff1]  = make_uint4(pack2bf(aRF[4].x, aRF[4].y), pack2bf(aRF[4].z, aRF[4].w),
                                                  pack2bf(aRF[5].x, aRF[5].y), pack2bf(aRF[5].z, aRF[5].w));
                *(uint4*)&Ad[aOff1b] = make_uint4(pack2bf(aRF[6].x, aRF[6].y), pack2bf(aRF[6].z, aRF[6].w),
                                                  pack2bf(aRF[7].x, aRF[7].y), pack2bf(aRF[7].z, aRF[7].w));
            }
            *(uint4*)&Bd[bOff] = bR;

            if (k0 + TK < H) {                  // preload next slice (hides HBM latency)
                if constexpr (PRECONV) {
                    const size_t kd = (size_t)((k0 + TK) >> 1);
                    aRB[0] = *(const uint4*)(xprevB + aGI0 + kd);
                    aRB[1] = *(const uint4*)(xprevB + aGI0 + kd + 4);
                    aRB[2] = *(const uint4*)(xprevB + aGI1 + kd);
                    aRB[3] = *(const uint4*)(xprevB + aGI1 + kd + 4);
                } else {
                    const float4* a0 = (const float4*)(xprevF + (size_t)(mBase + arow) * H + ahalf * 16 + k0 + TK);
                    const float4* a1 = (const float4*)(xprevF + (size_t)(mBase + arow + 128) * H + ahalf * 16 + k0 + TK);
#pragma unroll
                    for (int q = 0; q < 4; ++q) { aRF[q] = a0[q]; aRF[4 + q] = a1[q]; }
                }
                bR = *(const uint4*)(Wb + bGI + ((k0 + TK) >> 1));
            }
            __syncthreads();
            compute(Ad, Bd);
        }
    }

    // ---- Epilogue: D layout -> VGPR r: M = r + 8*half, N = l15 ----
#pragma unroll
    for (int mi = 0; mi < 2; ++mi) {
#pragma unroll
        for (int s = 0; s < 4; ++s) {
            const int col  = nBase + s * 16 + l15;
            const int rw0  = mBase + wave * 32 + mi * 16 + half * 8;
            const size_t base = (size_t)rw0 * H + col;
            float xc[8];
#pragma unroll
            for (int r = 0; r < 8; ++r) xc[r] = xcur[base + (size_t)r * H];
#pragma unroll
            for (int r = 0; r < 8; ++r)
                out[base + (size_t)r * H] = xc[r] + alpha * acc[mi][s][r];
        }
    }
}

extern "C" void kernel_launch(void* const* d_in, const int* in_sizes, int n_in,
                              void* d_out, int out_size, void* d_ws, size_t ws_size,
                              hipStream_t stream) {
    const float* xcur   = (const float*)d_in[0];   // (B,S,H) f32
    const float* xprev  = (const float*)d_in[1];   // (B,S,H) f32
    const float* alphap = (const float*)d_in[2];   // scalar f32
    const float* vals   = (const float*)d_in[3];   // (NNZ,) f32
    const int*   idx    = (const int*)d_in[4];     // (2,NNZ) i32
    float*       out    = (float*)d_out;

    const int nnz  = in_sizes[3];
    const int rows = in_sizes[0] / H;              // 8192

    const size_t wfBytes = (size_t)H * H * 4;      // 16 MB
    const size_t wbBytes = (size_t)H * H * 2;      //  8 MB
    const size_t xbBytes = (size_t)rows * H * 2;   // 32 MB

    float*        Wf = (float*)d_ws;
    unsigned int* Wb = (unsigned int*)((char*)d_ws + wfBytes);
    unsigned int* Xb = (unsigned int*)((char*)d_ws + wfBytes + wbBytes);

    cl_zero_kernel<<<(H * H / 4) / 256, 256, 0, stream>>>((float4*)Wf);
    cl_scatter_kernel<<<(nnz + 255) / 256, 256, 0, stream>>>(idx, vals, Wf, nnz);
    cl_convert_kernel<<<(H * H / 4) / 256, 256, 0, stream>>>((const float4*)Wf, (uint2*)Wb);

    dim3 grid(H / TN, rows / BM);
    if (ws_size >= wfBytes + wbBytes + xbBytes) {
        cl_convert_kernel<<<((size_t)rows * H / 4) / 256, 256, 0, stream>>>(
            (const float4*)xprev, (uint2*)Xb);
        cl_gemm_kernel<true><<<grid, 256, 0, stream>>>(nullptr, Xb, xcur, Wb, alphap, out);
    } else {
        cl_gemm_kernel<false><<<grid, 256, 0, stream>>>(xprev, nullptr, xcur, Wb, alphap, out);
    }
}
